// SAVSR_47648367182198
// MI455X (gfx1250) — matrix operand order repaired and checked by reading, not compiled, not run on hardware
//
#include <hip/hip_runtime.h>
#include <hip/hip_bf16.h>

// ---------------------------------------------------------------------------
// SAVSR dynamic-conv for MI455X (gfx1250, wave32, WMMA + async/TDM staging).
//
// Pipeline:
//   1) gap_kernel      : per-(b,c) mean over 160x160
//   2) att_kernel      : routing MLP + BN + 4 attention heads (1 block)
//   3) aggw_kernel     : aggregate expert weights -> f16 A-matrix [b][o][K=576]
//                        with K ordered tap-major (k = tap*64 + c)
//   4) xc_kernel       : xc = x * channel_att, transposed to [b][y][x][c] f16
//   5) conv_wmma_kernel: implicit GEMM, v_wmma_f32_16x16x32_f16
//        - weights staged via TDM tensor_load_to_lds (TENSORcnt)
//        - x tile staged via global_load_async_to_lds_b128 (ASYNCcnt)
//        - 2 N-tiles per K-step sharing one A fragment
//
// ws layout (needs ~53.7 MB; out itself is 105 MB so ws is assumed >= that):
//   [0)            xc_f16   16*160*160*64*2 = 52,428,800 B
//   [52428800)     aggw_f16 16*64*576*2     =  1,179,648 B
//   [53608448)     gap      1024 f32
//   [+4096)        ch_att   1024 f32
//   [+4096)        f_att    1024 f32
//   [+4096)        sp_att   144  f32
//   [+576)         k_att    128  f32
// ---------------------------------------------------------------------------

typedef __attribute__((ext_vector_type(16))) _Float16 v16h;
typedef __attribute__((ext_vector_type(8)))  float    v8f;
typedef __attribute__((ext_vector_type(4)))  unsigned int v4u;
typedef __attribute__((ext_vector_type(8)))  int      v8i;
typedef __attribute__((ext_vector_type(4)))  int      v4i;

#define BB 16
#define CC 64
#define HH 160
#define WW 160
#define KK 8
#define OO 64
#define AA 16
#define KDIM 576              // 9 taps * 64 channels
#define ROWS_PER_BLOCK 2
#define XL_ROWS 4             // ROWS_PER_BLOCK + 2 halo
#define XL_COLS 162           // WW + 2 halo

// ---------------------------------------------------------------- 1) GAP ----
__global__ void gap_kernel(const float* __restrict__ x, float* __restrict__ gap) {
  __shared__ float red[256];
  int bc = blockIdx.x;                       // (b*64 + c), 1024 blocks
  const float* p = x + (size_t)bc * (HH * WW);
  float s = 0.f;
  for (int i = threadIdx.x; i < HH * WW; i += 256) s += p[i];
  red[threadIdx.x] = s;
  __syncthreads();
  for (int off = 128; off > 0; off >>= 1) {
    if ((int)threadIdx.x < off) red[threadIdx.x] += red[threadIdx.x + off];
    __syncthreads();
  }
  if (threadIdx.x == 0) gap[bc] = red[0] * (1.0f / (HH * WW));
}

// ------------------------------------------------------- 2) attention MLP ---
__global__ void att_kernel(const float* __restrict__ gap, const float* __restrict__ scale,
                           const float* __restrict__ rw1, const float* __restrict__ rb1,
                           const float* __restrict__ rw2, const float* __restrict__ rb2,
                           const float* __restrict__ fc_w,
                           const float* __restrict__ bn_g, const float* __restrict__ bn_b,
                           const float* __restrict__ bn_m, const float* __restrict__ bn_v,
                           const float* __restrict__ cfc_w, const float* __restrict__ cfc_b,
                           const float* __restrict__ ffc_w, const float* __restrict__ ffc_b,
                           const float* __restrict__ sfc_w, const float* __restrict__ sfc_b,
                           const float* __restrict__ kfc_w, const float* __restrict__ kfc_b,
                           float* __restrict__ ch_att, float* __restrict__ f_att,
                           float* __restrict__ sp_att, float* __restrict__ k_att) {
  __shared__ float s[BB][CC + 2];
  __shared__ float h[BB][2 * CC];
  __shared__ float routed[BB][CC];
  __shared__ float v[BB][AA];
  int tid = threadIdx.x;

  for (int i = tid; i < BB * (CC + 2); i += 256) {
    int b = i / (CC + 2), j = i % (CC + 2);
    s[b][j] = (j < 2) ? (1.0f / scale[j]) : gap[b * CC + (j - 2)];
  }
  __syncthreads();
  for (int i = tid; i < BB * 2 * CC; i += 256) {          // h = relu(s @ rw1^T + rb1)
    int b = i / (2 * CC), j = i % (2 * CC);
    float acc = rb1[j];
    for (int k = 0; k < CC + 2; ++k) acc += s[b][k] * rw1[j * (CC + 2) + k];
    h[b][j] = fmaxf(acc, 0.f);
  }
  __syncthreads();
  for (int i = tid; i < BB * CC; i += 256) {              // routed = relu(h @ rw2^T + rb2)
    int b = i / CC, j = i % CC;
    float acc = rb2[j];
    for (int k = 0; k < 2 * CC; ++k) acc += h[b][k] * rw2[j * 2 * CC + k];
    routed[b][j] = fmaxf(acc, 0.f);
  }
  __syncthreads();
  for (int i = tid; i < BB * AA; i += 256) {              // v = relu(BN(routed @ fc_w^T))
    int b = i / AA, j = i % AA;
    float acc = 0.f;
    for (int k = 0; k < CC; ++k) acc += routed[b][k] * fc_w[j * CC + k];
    acc = (acc - bn_m[j]) * rsqrtf(bn_v[j] + 1e-5f) * bn_g[j] + bn_b[j];
    v[b][j] = fmaxf(acc, 0.f);
  }
  __syncthreads();
  for (int i = tid; i < BB * CC; i += 256) {              // channel / filter heads
    int b = i / CC, j = i % CC;
    float a1 = cfc_b[j], a2 = ffc_b[j];
    for (int k = 0; k < AA; ++k) {
      a1 += v[b][k] * cfc_w[j * AA + k];
      a2 += v[b][k] * ffc_w[j * AA + k];
    }
    ch_att[i] = 1.f / (1.f + __expf(-a1));
    f_att[i]  = 1.f / (1.f + __expf(-a2));
  }
  for (int i = tid; i < BB * 9; i += 256) {               // spatial head
    int b = i / 9, j = i % 9;
    float a = sfc_b[j];
    for (int k = 0; k < AA; ++k) a += v[b][k] * sfc_w[j * AA + k];
    sp_att[i] = 1.f / (1.f + __expf(-a));
  }
  if (tid < BB) {                                         // kernel head: softmax over K=8
    int b = tid;
    float lg[KK];
    float mx = -1e30f;
    for (int j = 0; j < KK; ++j) {
      float a = kfc_b[j];
      for (int k = 0; k < AA; ++k) a += v[b][k] * kfc_w[j * AA + k];
      lg[j] = a;
      mx = fmaxf(mx, a);
    }
    float sum = 0.f;
    for (int j = 0; j < KK; ++j) { lg[j] = __expf(lg[j] - mx); sum += lg[j]; }
    for (int j = 0; j < KK; ++j) k_att[b * KK + j] = lg[j] / sum;
  }
}

// ------------------------------------------- 3) aggregate expert weights ----
__global__ void aggw_kernel(const float* __restrict__ weight, const float* __restrict__ sp_att,
                            const float* __restrict__ k_att, _Float16* __restrict__ aggw) {
  __shared__ float wo[KK][CC * 9];                        // 18 KB: this o's experts
  int o = blockIdx.x;                                     // 64 blocks
  for (int i = threadIdx.x; i < KK * CC * 9; i += 256) {
    int k = i / (CC * 9), r = i % (CC * 9);
    wo[k][r] = weight[((size_t)k * OO + o) * (CC * 9) + r];   // [e][o][c][t] contiguous
  }
  __syncthreads();
  for (int b = 0; b < BB; ++b) {
    float ka[KK];
    for (int k = 0; k < KK; ++k) ka[k] = k_att[b * KK + k];
    for (int idx = threadIdx.x; idx < KDIM; idx += 256) {
      int t = idx >> 6, c = idx & 63;
      float acc = 0.f;
      for (int k = 0; k < KK; ++k) acc += ka[k] * wo[k][c * 9 + t];
      acc *= sp_att[b * 9 + t];
      aggw[((size_t)(b * OO + o)) * KDIM + idx] = (_Float16)acc;
    }
  }
}

// ----------------------- 4) gated input, transposed to [b][y][x][c] f16 -----
__global__ void xc_kernel(const float* __restrict__ x, const float* __restrict__ ch_att,
                          _Float16* __restrict__ xc) {
  __shared__ _Float16 tile[WW][72];                       // pad 64->72 (36-dword row stride)
  int b = blockIdx.x / HH, y = blockIdx.x % HH;           // 2560 blocks
  for (int i = threadIdx.x; i < CC * WW; i += 256) {      // coalesced read along x
    int c = i / WW, xcol = i % WW;
    float val = x[(((size_t)b * CC + c) * HH + y) * WW + xcol] * ch_att[b * CC + c];
    tile[xcol][c] = (_Float16)val;
  }
  __syncthreads();
  for (int i = threadIdx.x; i < WW * CC / 8; i += 256) {  // 16B-coalesced write along c
    int xcol = i / 8, c8 = i % 8;
    union { uint4 q; _Float16 hh[8]; } u;
    for (int j = 0; j < 8; ++j) u.hh[j] = tile[xcol][c8 * 8 + j];
    *(uint4*)(xc + ((((size_t)b * HH + y) * WW + xcol) * CC) + c8 * 8) = u.q;
  }
}

// --------------------------------------------- 5) implicit-GEMM WMMA conv --
// Block: 1 batch x 2 output rows x 160 cols x 64 out-channels. 8 waves:
//   wave = (mo in 0..3) x (row r in 0..1); each wave: 5 tile-pairs x 18 K-steps.
__global__ __launch_bounds__(256, 1)
void conv_wmma_kernel(const _Float16* __restrict__ xc, const _Float16* __restrict__ aggw,
                      const float* __restrict__ f_att, float* __restrict__ out) {
  __shared__ _Float16 wlds[OO * KDIM];                    // 73,728 B: A [o][k]
  __shared__ _Float16 xlds[XL_ROWS * XL_COLS * CC];       // 82,944 B: B [ry][cx][c]

  int blk = blockIdx.x;                                   // 16 * 80 = 1280 blocks
  int b = blk / (HH / ROWS_PER_BLOCK);
  int y0 = (blk % (HH / ROWS_PER_BLOCK)) * ROWS_PER_BLOCK;
  int tid = threadIdx.x;
  int wave = tid >> 5, lane = tid & 31;

  // --- stage this batch's aggregated weights via the Tensor Data Mover -----
  // 1-D tensor, 2-byte elements, 36864 of them (73,728 B), one descriptor.
#if __has_builtin(__builtin_amdgcn_tensor_load_to_lds)
  if (wave == 0) {
    unsigned long long ga =
        (unsigned long long)(uintptr_t)(aggw + (size_t)b * OO * KDIM);
    unsigned int lds = (unsigned int)(uintptr_t)wlds;     // flat low 32b = LDS offset
    const unsigned int NH = OO * KDIM;                    // 36864 elements
    v4u g0;
    g0[0] = 1u;                                           // count=1 (user mode)
    g0[1] = lds;                                          // [63:32]  lds_addr
    g0[2] = (unsigned int)ga;                             // [95:64]  global_addr lo
    g0[3] = (unsigned int)((ga >> 32) & 0x01FFFFFFu) | (2u << 30); // addr hi | type=2
    v8i g1;
    g1[0] = (int)(1u << 16);                              // data_size=1 (2 bytes)
    g1[1] = (int)((NH & 0xFFFFu) << 16);                  // [63:48] tensor_dim0 lo16
    g1[2] = (int)((NH >> 16) | (1u << 16));               // tensor_dim0 hi16 | dim1=1
    g1[3] = (int)(NH << 16);                              // [127:112] tile_dim0
    g1[4] = 1;                                            // tile_dim1=1, tile_dim2=0
    g1[5] = (int)NH;                                      // tensor_dim0_stride lo32
    g1[6] = 0;
    g1[7] = 0;
    v4i z4 = {0, 0, 0, 0};
#if defined(__clang_major__) && (__clang_major__ >= 23)
    v8i z8 = {0, 0, 0, 0, 0, 0, 0, 0};
    __builtin_amdgcn_tensor_load_to_lds(g0, g1, z4, z4, z8, 0);
#else
    __builtin_amdgcn_tensor_load_to_lds(g0, g1, z4, z4, 0);
#endif
    __builtin_amdgcn_s_wait_tensorcnt(0);
  }
#else
  {
    const uint4* src = (const uint4*)(aggw + (size_t)b * OO * KDIM);
    uint4* dst = (uint4*)wlds;
    for (int i = tid; i < OO * KDIM / 8; i += 256) dst[i] = src[i];
  }
#endif

  // --- stage halo'd x tile: async global->LDS for in-range cells, ----------
  // --- plain ds zero-stores for the padding cells (disjoint addresses) -----
  {
    const int row_q = XL_COLS * CC / 8;                   // 1296 uint4 per ry
    for (int i = tid; i < XL_ROWS * row_q; i += 256) {
      int ry = i / row_q, rem = i % row_q;
      int cx = rem / (CC / 8), c8 = rem % (CC / 8);
      int gy = y0 - 1 + ry, gx = cx - 1;
      unsigned int lds_off = (unsigned int)(uintptr_t)xlds + (unsigned int)i * 16u;
      if ((unsigned)gy < (unsigned)HH && (unsigned)gx < (unsigned)WW) {
        unsigned long long ga = (unsigned long long)(uintptr_t)
            (xc + ((((size_t)b * HH + gy) * WW + gx) * CC) + c8 * 8);
        asm volatile("global_load_async_to_lds_b128 %0, %1, off"
                     :: "v"(lds_off), "v"(ga) : "memory");
      } else {
        uint4 z = {0u, 0u, 0u, 0u};
        ((uint4*)xlds)[i] = z;
      }
    }
    asm volatile("s_wait_asynccnt 0x0" ::: "memory");
  }
  __syncthreads();

  int mo = wave & 3;                                      // 16-row M tile
  int r  = wave >> 2;                                     // output row in chunk
  int y  = y0 + r;
  int px = lane & 15;                                     // N column within tile
  bool lo = lane < 16;

  float fa[8];                                            // filter_att per C-frag row
  for (int rr = 0; rr < 8; ++rr)
    fa[rr] = f_att[b * OO + mo * 16 + rr + (lo ? 0 : 8)];

  // A layout: lane m-row = mo*16 + (lane&15); low lanes K{0-15}, high K{16-31}
  const int aoff0 = (mo * 16 + (lane & 15)) * KDIM + (lo ? 0 : 16);
  // B layout: low lanes K{0-15}, high lanes K{16-31}, column = px
  const int bsub = lo ? 0 : 16;

  union AB { v16h v; uint4 q[2]; };

  for (int jp = 0; jp < 5; ++jp) {                        // pairs of 16-px tiles
    int x0 = jp * 32;
    v8f acc0 = {}, acc1 = {};
#pragma unroll
    for (int kk = 0; kk < KDIM; kk += 32) {
      int t = kk >> 6;                                    // tap 0..8
      int c0 = kk & 63;                                   // 0 or 32
      int dy = t / 3, dx = t % 3;
      AB a, b0, b1;
      const uint4* ap = (const uint4*)(wlds + aoff0 + kk);
      a.q[0] = ap[0];                                     // K {+0..7}
      a.q[1] = ap[1];                                     // K {+8..15}
      int base = ((r + dy) * XL_COLS + (x0 + dx + px)) * CC + c0 + bsub;
      const uint4* bp0 = (const uint4*)(xlds + base);
      const uint4* bp1 = (const uint4*)(xlds + base + 16 * CC);
      b0.q[0] = bp0[0]; b0.q[1] = bp0[1];
      b1.q[0] = bp1[0]; b1.q[1] = bp1[1];
      acc0 = __builtin_amdgcn_wmma_f32_16x16x32_f16(false, a.v, false, b0.v,
                                                    (short)0, acc0, false, false);
      acc1 = __builtin_amdgcn_wmma_f32_16x16x32_f16(false, a.v, false, b1.v,
                                                    (short)0, acc1, false, false);
    }
#pragma unroll
    for (int rr = 0; rr < 8; ++rr) {
      int m = mo * 16 + rr + (lo ? 0 : 8);
      size_t o0 = (((size_t)(b * OO + m)) * HH + y) * WW + x0 + px;
      out[o0]      = acc0[rr] * fa[rr];
      out[o0 + 16] = acc1[rr] * fa[rr];
    }
  }
}

// ---------------------------------------------------------------------------
extern "C" void kernel_launch(void* const* d_in, const int* in_sizes, int n_in,
                              void* d_out, int out_size, void* d_ws, size_t ws_size,
                              hipStream_t stream) {
  const float* x      = (const float*)d_in[0];
  const float* scale  = (const float*)d_in[1];
  const float* rw1    = (const float*)d_in[2];
  const float* rb1    = (const float*)d_in[3];
  const float* rw2    = (const float*)d_in[4];
  const float* rb2    = (const float*)d_in[5];
  const float* fc_w   = (const float*)d_in[6];
  const float* bn_g   = (const float*)d_in[7];
  const float* bn_b   = (const float*)d_in[8];
  const float* bn_m   = (const float*)d_in[9];
  const float* bn_v   = (const float*)d_in[10];
  const float* cfc_w  = (const float*)d_in[11];
  const float* cfc_b  = (const float*)d_in[12];
  const float* ffc_w  = (const float*)d_in[13];
  const float* ffc_b  = (const float*)d_in[14];
  const float* sfc_w  = (const float*)d_in[15];
  const float* sfc_b  = (const float*)d_in[16];
  const float* kfc_w  = (const float*)d_in[17];
  const float* kfc_b  = (const float*)d_in[18];
  const float* weight = (const float*)d_in[19];

  char* ws = (char*)d_ws;
  _Float16* xc   = (_Float16*)ws;                         // 52,428,800 B
  _Float16* aggw = (_Float16*)(ws + 52428800);            //  1,179,648 B
  float* gap    = (float*)(ws + 53608448);
  float* ch_att = gap + 1024;
  float* f_att  = ch_att + 1024;
  float* sp_att = f_att + 1024;
  float* k_att  = sp_att + 144;

  gap_kernel<<<BB * CC, 256, 0, stream>>>(x, gap);
  att_kernel<<<1, 256, 0, stream>>>(gap, scale, rw1, rb1, rw2, rb2, fc_w,
                                    bn_g, bn_b, bn_m, bn_v, cfc_w, cfc_b,
                                    ffc_w, ffc_b, sfc_w, sfc_b, kfc_w, kfc_b,
                                    ch_att, f_att, sp_att, k_att);
  aggw_kernel<<<OO, 256, 0, stream>>>(weight, sp_att, k_att, aggw);
  xc_kernel<<<BB * HH, 256, 0, stream>>>(x, ch_att, xc);
  conv_wmma_kernel<<<BB * (HH / ROWS_PER_BLOCK), 256, 0, stream>>>(xc, aggw, f_att,
                                                                   (float*)d_out);
}
